// MultiHeadAttention_52613349376273
// MI455X (gfx1250) — matrix operand order repaired and checked by reading, not compiled, not run on hardware
//
#include <hip/hip_runtime.h>

// MI455X / gfx1250 multi-head attention, bf16 WMMA + async-to-LDS path.
// B=2 S=2048 D=1024 H=16 DK=64. Causal mask implemented analytically.
#define B_  2
#define S_  2048
#define D_  1024
#define H_  16
#define DK_ 64
#define M_  (B_ * S_) // 4096 rows in all projection GEMMs

typedef __attribute__((ext_vector_type(16))) __bf16 v16bf;
typedef __attribute__((ext_vector_type(8)))  float  v8f;
typedef __attribute__((ext_vector_type(4)))  int    v4i;
typedef __attribute__((address_space(1))) v4i* gptr4;  // global int4*
typedef __attribute__((address_space(3))) v4i* lptr4;  // LDS int4*

static __device__ __forceinline__ v8f wmma_bf16(v16bf a, v16bf b, v8f c) {
  // v_wmma_f32_16x16x32_bf16: D = A(16x32) * B(32x16) + C(16x16 f32)
  return __builtin_amdgcn_wmma_f32_16x16x32_bf16(false, a, false, b, (short)0, c,
                                                 false, false);
}

// ---- CDNA5 async copy: 16 bytes global -> LDS, tracked by ASYNCcnt ----------
static __device__ __forceinline__ void async_copy16(const __bf16* gsrc,
                                                    __bf16* lds_dst) {
#if __has_builtin(__builtin_amdgcn_global_load_async_to_lds_b128)
  __builtin_amdgcn_global_load_async_to_lds_b128(
      (gptr4)(void*)(const void*)gsrc, (lptr4)(void*)lds_dst, 0, 0);
#else
  // Fallback: synchronous 16B copy through VGPRs.
  *(ulonglong2*)lds_dst = *(const ulonglong2*)gsrc;
#endif
}

template <int N>
static __device__ __forceinline__ void wait_async() {
#if __has_builtin(__builtin_amdgcn_s_wait_asynccnt)
  __builtin_amdgcn_s_wait_asynccnt(N);
#else
  asm volatile("s_wait_asynccnt %0" ::"i"(N) : "memory");
#endif
}

// ---- Fragment loaders (ISA 7.12.2 layouts, wave32) --------------------------
// A 16x32 (MxK) bf16: lane<16 -> row=lane, K in {0..7,16..23};
//                     lane>=16 -> row=lane-16, K in {8..15,24..31};
//                     vgpr r holds K = ((r&4)<<2) + g*8 + 2*(r&3), +1
static __device__ __forceinline__ v16bf load_a_f32(const float* __restrict__ p,
                                                   int lda, int lane) {
  const int g = (lane >> 4) & 1, m = lane & 15;
  const float* row = p + (size_t)m * lda;
  v16bf a;
#pragma unroll
  for (int r = 0; r < 8; ++r) {
    const int k = ((r & 4) << 2) + g * 8 + 2 * (r & 3);
    a[2 * r]     = (__bf16)row[k];
    a[2 * r + 1] = (__bf16)row[k + 1];
  }
  return a;
}

static __device__ __forceinline__ v16bf load_a_bf16(const __bf16* __restrict__ p,
                                                    int lda, int lane) {
  const int g = (lane >> 4) & 1, m = lane & 15;
  const __bf16* row = p + (size_t)m * lda;
  v16bf a;
#pragma unroll
  for (int r = 0; r < 8; ++r) {
    const int k = ((r & 4) << 2) + g * 8 + 2 * (r & 3);
    a[2 * r]     = row[k];
    a[2 * r + 1] = row[k + 1];
  }
  return a;
}

// B 32x16 (KxN) bf16: lane = N (mod 16); lanes 0-15 hold K in {0..7,16..23},
// lanes 16-31 hold K in {8..15,24..31}; vgpr r holds K = ((r&4)<<2) + g*8 + 2*(r&3), +1.
static __device__ __forceinline__ v16bf load_b_f32(const float* __restrict__ p,
                                                   int ldb, int lane) {
  const int g = (lane >> 4) & 1, n = lane & 15;
  v16bf b;
#pragma unroll
  for (int r = 0; r < 8; ++r) {
    const int k = ((r & 4) << 2) + g * 8 + 2 * (r & 3);
    b[2 * r]     = (__bf16)p[(size_t)k * ldb + n];
    b[2 * r + 1] = (__bf16)p[(size_t)(k + 1) * ldb + n];
  }
  return b;
}

static __device__ __forceinline__ v16bf load_b_bf16(const __bf16* __restrict__ p,
                                                    int ldb, int lane) {
  const int g = (lane >> 4) & 1, n = lane & 15;
  v16bf b;
#pragma unroll
  for (int r = 0; r < 8; ++r) {
    const int k = ((r & 4) << 2) + g * 8 + 2 * (r & 3);
    b[2 * r]     = p[(size_t)k * ldb + n];
    b[2 * r + 1] = p[(size_t)(k + 1) * ldb + n];
  }
  return b;
}
// C 16x16 f32 frag: row m = r + 8*g, col n = lane&15.

// ---- Kernel 1: QKV projection, f32 in -> bf16 head-split layouts ------------
// Each wave: 32(M) x 64(N) tile -> every B fragment feeds 2 WMMAs.
// mode 0: Q [B,H,S,DK], mode 1: K^T [B,H,DK,S], mode 2: V [B,H,S,DK]
__global__ void __launch_bounds__(256) qkv_proj_kernel(
    const float* __restrict__ X, const float* __restrict__ W,
    __bf16* __restrict__ out, int mode) {
  const int lane = threadIdx.x & 31;
  const int wid  = blockIdx.x * 8 + (threadIdx.x >> 5); // 2048 wave-tasks
  const int nt = wid & 15;   // D/64 = 16 column super-tiles
  const int mt = wid >> 4;   // M/32 = 128 row tiles
  const int g = (lane >> 4) & 1, n = lane & 15;

  v8f c[2][4] = {};
  const float* xp = X + (size_t)mt * 32 * D_;
#pragma unroll 1
  for (int kk = 0; kk < D_; kk += 32) {
    v16bf a0 = load_a_f32(xp + kk,            D_, lane);
    v16bf a1 = load_a_f32(xp + 16 * D_ + kk,  D_, lane);
#pragma unroll
    for (int j = 0; j < 4; ++j) {
      v16bf b = load_b_f32(W + (size_t)kk * D_ + nt * 64 + j * 16, D_, lane);
      c[0][j] = wmma_bf16(a0, b, c[0][j]);
      c[1][j] = wmma_bf16(a1, b, c[1][j]);
    }
  }
#pragma unroll
  for (int i = 0; i < 2; ++i)
#pragma unroll
    for (int j = 0; j < 4; ++j)
#pragma unroll
      for (int r = 0; r < 8; ++r) {
        const int mg = mt * 32 + i * 16 + r + 8 * g;  // global row in [0, B*S)
        const int ng = nt * 64 + j * 16 + n;          // global col in [0, D)
        const int bb = mg >> 11, ss = mg & (S_ - 1);
        const int hh = ng >> 6,  dd = ng & (DK_ - 1);
        size_t idx;
        if (mode == 1)
          idx = (((size_t)(bb * H_ + hh)) * DK_ + dd) * S_ + ss; // K transposed
        else
          idx = (((size_t)(bb * H_ + hh)) * S_ + ss) * DK_ + dd; // Q / V
        out[idx] = (__bf16)c[i][j][r];
      }
}

// ---- Kernel 2: flash attention --------------------------------------------
// One wave per (b,h,16-query tile); per-wave double-buffered K^T/V tiles in
// LDS filled by GLOBAL_LOAD_ASYNC_TO_LDS_B128 (ASYNCcnt), overlapped with WMMA.
#define FA_WAVES 4
__global__ void __launch_bounds__(128) flash_attn_kernel(
    const __bf16* __restrict__ q, const __bf16* __restrict__ kt,
    const __bf16* __restrict__ v, __bf16* __restrict__ ocat) {
  constexpr int PSTR = 40;
  struct alignas(16) WaveBuf {
    __bf16 kt[2][64 * 32];  // K^T tile [d][s], 4KB per buffer
    __bf16 vt[2][32 * 64];  // V tile  [s][d], 4KB per buffer
    __bf16 p[16 * PSTR];    // P round-trip (C-layout -> A-layout)
  };
  __shared__ WaveBuf sb[FA_WAVES];  // ~70KB / block

  const int lane = threadIdx.x & 31;
  const int wave = threadIdx.x >> 5;
  const int wid  = blockIdx.x * FA_WAVES + wave;  // 4096 wave-tasks
  const int qtile = wid & 127;                    // S/16
  const int h     = (wid >> 7) & 15;
  const int b     = wid >> 11;
  const int g = (lane >> 4) & 1, n = lane & 15;
  const int q0 = qtile * 16;

  const __bf16* qp  = q  + (((size_t)(b * H_ + h)) * S_ + q0) * DK_; // [S,DK]
  const __bf16* ktp = kt + ((size_t)(b * H_ + h)) * DK_ * S_;        // [DK,S]
  const __bf16* vp0 = v  + ((size_t)(b * H_ + h)) * S_ * DK_;        // [S,DK]

  WaveBuf& wb = sb[wave];

  // Issue 16 async b128 copies (8 for K^T tile, 8 for V tile) for key block blk.
  auto issue_copies = [&](int blk) {
    const int kv0 = blk * 32;
    const int d   = blk & 1;
#pragma unroll
    for (int i = 0; i < 8; ++i) {          // K^T: 64 rows x 32 cols, 4 chunks/row
      const int ch = i * 32 + lane;
      async_copy16(ktp + (size_t)(ch >> 2) * S_ + kv0 + (ch & 3) * 8,
                   &wb.kt[d][ch * 8]);
    }
#pragma unroll
    for (int i = 0; i < 8; ++i) {          // V: 32 rows x 64 cols, 8 chunks/row
      const int ch = i * 32 + lane;
      async_copy16(vp0 + (size_t)(kv0 + (ch >> 3)) * DK_ + (ch & 7) * 8,
                   &wb.vt[d][ch * 8]);
    }
  };

  // Q tile as two A fragments (K = 0..31, 32..63), loaded once
  const v16bf aq0 = load_a_bf16(qp,      DK_, lane);
  const v16bf aq1 = load_a_bf16(qp + 32, DK_, lane);

  v8f o0 = {}, o1 = {}, o2 = {}, o3 = {};      // O accum 16x64 f32
  float mrow[8], lrow[8];                      // per-row running max / sum
#pragma unroll
  for (int r = 0; r < 8; ++r) { mrow[r] = -1e30f; lrow[r] = 0.f; }

  const int nblk = (qtile >> 1) + 1;           // causal: 32-wide key blocks
  issue_copies(0);

#pragma unroll 1
  for (int blk = 0; blk < nblk; ++blk) {
    const int kv0 = blk * 32;
    // Prefetch next block, then wait until the current buffer's 16 copies are
    // done (async ops complete in order: <=16 outstanding => first 16 landed).
    if (blk + 1 < nblk) { issue_copies(blk + 1); wait_async<16>(); }
    else                { wait_async<0>(); }

    const __bf16* kbuf = wb.kt[blk & 1];
    const __bf16* vbuf = wb.vt[blk & 1];

    // S = Q * K^T  (16x32 logits in two C fragments), B frags from LDS
    v8f s0 = {}, s1 = {};
    s0 = wmma_bf16(aq0, load_b_bf16(kbuf,            32, lane), s0);
    s0 = wmma_bf16(aq1, load_b_bf16(kbuf + 32 * 32,  32, lane), s0);
    s1 = wmma_bf16(aq0, load_b_bf16(kbuf + 16,       32, lane), s1);
    s1 = wmma_bf16(aq1, load_b_bf16(kbuf + 32 * 32 + 16, 32, lane), s1);

    // scale + causal mask + online softmax (row = r + 8g, col = kv0 + {n,16+n})
#pragma unroll
    for (int r = 0; r < 8; ++r) {
      const int qg = q0 + r + 8 * g;
      float x0 = s0[r] * 0.125f; if (kv0 + n      > qg) x0 += -1e9f;
      float x1 = s1[r] * 0.125f; if (kv0 + 16 + n > qg) x1 += -1e9f;
      float mx = fmaxf(x0, x1);
#pragma unroll
      for (int off = 8; off; off >>= 1) mx = fmaxf(mx, __shfl_xor(mx, off, 32));
      const float mnew  = fmaxf(mrow[r], mx);
      const float alpha = __expf(mrow[r] - mnew);
      mrow[r] = mnew;
      const float p0 = __expf(x0 - mnew);
      const float p1 = __expf(x1 - mnew);
      float rs = p0 + p1;
#pragma unroll
      for (int off = 8; off; off >>= 1) rs += __shfl_xor(rs, off, 32);
      lrow[r] = lrow[r] * alpha + rs;
      o0[r] *= alpha; o1[r] *= alpha; o2[r] *= alpha; o3[r] *= alpha;
      // spill P (C layout) row-major into this wave's LDS tile
      wb.p[(r + 8 * g) * PSTR + n]      = (__bf16)p0;
      wb.p[(r + 8 * g) * PSTR + 16 + n] = (__bf16)p1;
    }
    // LDS round-trip: wait for P stores (DS ops are wave-ordered) then reload
    // in A-fragment layout.
    asm volatile("s_wait_dscnt 0" ::: "memory");
    const v16bf ap = load_a_bf16(wb.p, PSTR, lane);

    // O += P * V   (K = 32 key rows, N = 64 head dims), B frags from LDS
    o0 = wmma_bf16(ap, load_b_bf16(vbuf +  0, 64, lane), o0);
    o1 = wmma_bf16(ap, load_b_bf16(vbuf + 16, 64, lane), o1);
    o2 = wmma_bf16(ap, load_b_bf16(vbuf + 32, 64, lane), o2);
    o3 = wmma_bf16(ap, load_b_bf16(vbuf + 48, 64, lane), o3);
  }

  // normalize and write concat-head output [B,S,D] bf16
#pragma unroll
  for (int r = 0; r < 8; ++r) {
    const float inv = 1.0f / lrow[r];
    const int srow = q0 + r + 8 * g;
    __bf16* orow = ocat + ((size_t)(b * S_ + srow)) * D_ + h * DK_;
    orow[ 0 + n] = (__bf16)(o0[r] * inv);
    orow[16 + n] = (__bf16)(o1[r] * inv);
    orow[32 + n] = (__bf16)(o2[r] * inv);
    orow[48 + n] = (__bf16)(o3[r] * inv);
  }
}

// ---- Kernel 3: output projection, bf16 A x f32 W -> f32 out -----------------
__global__ void __launch_bounds__(256) out_proj_kernel(
    const __bf16* __restrict__ A, const float* __restrict__ W,
    float* __restrict__ out) {
  const int lane = threadIdx.x & 31;
  const int wid  = blockIdx.x * 8 + (threadIdx.x >> 5);
  const int nt = wid & 15, mt = wid >> 4;   // 32x64 tile per wave
  const int g = (lane >> 4) & 1, n = lane & 15;

  v8f c[2][4] = {};
  const __bf16* ap = A + (size_t)mt * 32 * D_;
#pragma unroll 1
  for (int kk = 0; kk < D_; kk += 32) {
    v16bf a0 = load_a_bf16(ap + kk,           D_, lane);
    v16bf a1 = load_a_bf16(ap + 16 * D_ + kk, D_, lane);
#pragma unroll
    for (int j = 0; j < 4; ++j) {
      v16bf b = load_b_f32(W + (size_t)kk * D_ + nt * 64 + j * 16, D_, lane);
      c[0][j] = wmma_bf16(a0, b, c[0][j]);
      c[1][j] = wmma_bf16(a1, b, c[1][j]);
    }
  }
#pragma unroll
  for (int i = 0; i < 2; ++i)
#pragma unroll
    for (int j = 0; j < 4; ++j)
#pragma unroll
      for (int r = 0; r < 8; ++r)
        out[((size_t)(mt * 32 + i * 16 + r + 8 * g)) * D_ + nt * 64 + j * 16 + n]
            = c[i][j][r];
}

// ---- Host entry -------------------------------------------------------------
extern "C" void kernel_launch(void* const* d_in, const int* in_sizes, int n_in,
                              void* d_out, int out_size, void* d_ws, size_t ws_size,
                              hipStream_t stream) {
  const float* x  = (const float*)d_in[0];
  // d_in[1] is the causal mask; it is exactly triu(1)*(-1e9), applied analytically.
  const float* wq = (const float*)d_in[2];
  const float* wk = (const float*)d_in[3];
  const float* wv = (const float*)d_in[4];
  const float* wo = (const float*)d_in[5];
  float* out = (float*)d_out;

  const size_t NPH = (size_t)B_ * H_ * S_ * DK_;  // 4,194,304 elems per tensor
  __bf16* qb  = (__bf16*)d_ws;   // 8 MB each; total 32 MB of workspace
  __bf16* ktb = qb + NPH;
  __bf16* vb  = ktb + NPH;
  __bf16* ob  = vb + NPH;

  qkv_proj_kernel<<<dim3(256), dim3(256), 0, stream>>>(x, wq, qb, 0);
  qkv_proj_kernel<<<dim3(256), dim3(256), 0, stream>>>(x, wk, ktb, 1);
  qkv_proj_kernel<<<dim3(256), dim3(256), 0, stream>>>(x, wv, vb, 2);
  flash_attn_kernel<<<dim3(1024), dim3(128), 0, stream>>>(qb, ktb, vb, ob);
  out_proj_kernel<<<dim3(256), dim3(256), 0, stream>>>(ob, wo, out);

  (void)in_sizes; (void)n_in; (void)out_size; (void)ws_size;
}
